// RelativeMultiHeadAttentionLayer_17806934409471
// MI455X (gfx1250) — matrix operand order repaired and checked by reading, not compiled, not run on hardware
//
#include <hip/hip_runtime.h>

typedef _Float16 h16;
typedef __attribute__((ext_vector_type(16))) _Float16 v16h;
typedef __attribute__((ext_vector_type(8)))  float    v8f;

#define WMMA_F16(a, b, c) \
    __builtin_amdgcn_wmma_f32_16x16x32_f16(false, (a), false, (b), (short)0, (c), false, false)

// ---------------------------------------------------------------------------
// Fragment loaders per CDNA5 ISA VGPR layouts (CDNA5 ISA).
// A 16x32 f16: lanes 0-15 (M=lane): K = [0..7]+[16..23]; lanes 16-31: K=[8..15]+[24..31]
// B 32x16 f16: lanes 0-15 (N=lane): K = [0..7]+[16..23]; lanes 16-31: K=[8..15]+[24..31] (same as A).
// rowPtr points at the 32-half K-chunk for this lane's row. hi = lane>>4.
// ---------------------------------------------------------------------------
__device__ __forceinline__ v16h load_frag_A(const h16* rowPtr, int hi) {
    union { v16h v; uint4 u[2]; } r;
    r.u[0] = *(const uint4*)(rowPtr + hi * 8);
    r.u[1] = *(const uint4*)(rowPtr + 16 + hi * 8);
    return r.v;
}
__device__ __forceinline__ v16h load_frag_B(const h16* rowPtr, int hi) {
    union { v16h v; uint4 u[2]; } r;
    r.u[0] = *(const uint4*)(rowPtr + hi * 8);
    r.u[1] = *(const uint4*)(rowPtr + 16 + hi * 8);
    return r.v;
}

// ---------------------------------------------------------------------------
// fp32 -> fp16 conversion, 8 elements/thread (full-rate HBM streaming)
// ---------------------------------------------------------------------------
__global__ void f32_to_f16_kernel(const float* __restrict__ in, h16* __restrict__ out, int n) {
    int i = (blockIdx.x * blockDim.x + threadIdx.x) * 8;
    if (i < n) {
        float4 a = *(const float4*)(in + i);
        float4 b = *(const float4*)(in + i + 4);
        h16 tmp[8] = {(h16)a.x, (h16)a.y, (h16)a.z, (h16)a.w,
                      (h16)b.x, (h16)b.y, (h16)b.z, (h16)b.w};
        *(uint4*)(out + i) = *(const uint4*)tmp;
    }
}

// ---------------------------------------------------------------------------
// C[m][n] = sum_k A[m][k] * Bm[n][k] + bias[n]        (i.e. A @ W^T + b)
// A : M x K f16 row-major.   Bm : N x K f16 row-major (== original W).
// mode 0: write f16 head-major out[((b*16+h)*512+s)*64+d] * outScale
// mode 1: write f32 row-major outF[m*N+n]
// Block: 256 thr (8 waves), 128x128 tile, K chunks of 32, double-buffered LDS.
// Wave grid 4x2: wave (wr,wc) computes rows wr*32..+31, cols wc*64..+63
//   -> 2x4 = 8 WMMA tiles per wave per k-chunk.
// ---------------------------------------------------------------------------
__global__ __launch_bounds__(256) void gemm_xwT_kernel(
    const h16* __restrict__ A, const h16* __restrict__ Bm,
    const float* __restrict__ bias,
    h16* __restrict__ outH, float* __restrict__ outF,
    int M, int N, int K, int mode, float outScale)
{
    __shared__ __align__(16) h16 As[2][128][32];
    __shared__ __align__(16) h16 Bs[2][128][32];

    const int tid = threadIdx.x;
    const int w   = tid >> 5;          // 0..7
    const int l   = tid & 31;
    const int lo  = l & 15;
    const int hi  = l >> 4;
    const int wr  = w >> 1;            // 0..3 : 32-row strip
    const int wc  = w & 1;             // 0..1 : 64-col strip
    const int m0  = blockIdx.y * 128;
    const int n0  = blockIdx.x * 128;

    const int row = tid >> 1;          // 0..127, two threads per LDS row
    const int off = (tid & 1) * 16;    // halves

    const h16* gA = A  + (size_t)(m0 + row) * K + off;
    const h16* gB = Bm + (size_t)(n0 + row) * K + off;

    auto loadTile = [&](int buf, int kc) {
        const uint4* ga = (const uint4*)(gA + kc);
        uint4* sa = (uint4*)(&As[buf][row][off]);
        sa[0] = ga[0]; sa[1] = ga[1];
        const uint4* gb = (const uint4*)(gB + kc);
        uint4* sb = (uint4*)(&Bs[buf][row][off]);
        sb[0] = gb[0]; sb[1] = gb[1];
    };

    v8f acc[2][4] = {};

    loadTile(0, 0);
    __syncthreads();
    int cur = 0;
    for (int kc = 0; kc < K; kc += 32) {
        if (kc + 32 < K) loadTile(cur ^ 1, kc + 32);
        if (kc + 64 < K) {                       // global_prefetch for chunk k+2
            __builtin_prefetch(gA + kc + 64, 0, 3);
            __builtin_prefetch(gB + kc + 64, 0, 3);
        }
        v16h af[2], bf[4];
#pragma unroll
        for (int i = 0; i < 2; ++i)
            af[i] = load_frag_A(&As[cur][wr * 32 + i * 16 + lo][0], hi);
#pragma unroll
        for (int j = 0; j < 4; ++j)
            bf[j] = load_frag_B(&Bs[cur][wc * 64 + j * 16 + lo][0], hi);
#pragma unroll
        for (int i = 0; i < 2; ++i)
#pragma unroll
            for (int j = 0; j < 4; ++j)
                acc[i][j] = WMMA_F16(af[i], bf[j], acc[i][j]);
        __syncthreads();
        cur ^= 1;
    }

    // Epilogue. C layout: VGPR r, lanes 0-15 -> (M=r, N=lane); lanes 16-31 -> (M=8+r).
#pragma unroll
    for (int i = 0; i < 2; ++i) {
#pragma unroll
        for (int j = 0; j < 4; ++j) {
            const int gn = n0 + wc * 64 + j * 16 + lo;
            const float bv = bias[gn];
#pragma unroll
            for (int r = 0; r < 8; ++r) {
                const int gm = m0 + wr * 32 + i * 16 + r + 8 * hi;
                float val = (acc[i][j][r] + bv) * outScale;
                if (mode == 0) {
                    int b = gm >> 9, s = gm & 511;       // S = 512
                    int h = gn >> 6, d = gn & 63;        // dk = 64
                    outH[(((size_t)(b * 16 + h) * 512) + s) * 64 + d] = (h16)val;
                } else {
                    outF[(size_t)gm * N + gn] = val;
                }
            }
        }
    }
}

// ---------------------------------------------------------------------------
// Flash attention per (query-block 64, head, batch). 4 waves, wave = 16 q rows.
// q already scaled by 1/sqrt(dk). Bias: rel_table[i + 511 - j][h].
// Output f16 row-major ahat[b*512+s][h*64+d].
// ---------------------------------------------------------------------------
__global__ __launch_bounds__(128) void attn_kernel(
    const h16* __restrict__ q, const h16* __restrict__ k, const h16* __restrict__ v,
    const float* __restrict__ rel_table, h16* __restrict__ outA)
{
    const int S = 512, DK = 64, H = 16;
    __shared__ __align__(16) h16 Ks[64][64];     // Ks[key][d]
    __shared__ __align__(16) h16 Vt[64][64];     // Vt[d][key]  (transposed)
    __shared__ __align__(16) h16 Ps[4][16][64];  // per-wave P staging
    __shared__ float sBias[128];

    const int qb = blockIdx.x * 64;
    const int h  = blockIdx.y;
    const int b  = blockIdx.z;
    const int tid = threadIdx.x;
    const int w  = tid >> 5, l = tid & 31, lo = l & 15, hi = l >> 4;

    const size_t headOff = ((size_t)(b * H + h)) * S * DK;
    const h16* qp = q + headOff;
    const h16* kp = k + headOff;
    const h16* vp = v + headOff;

    // Q fragments for this wave's 16 rows (direct from global, A layout).
    v16h qf[2];
#pragma unroll
    for (int c = 0; c < 2; ++c)
        qf[c] = load_frag_A(qp + (size_t)(qb + w * 16 + lo) * DK + c * 32, hi);

    v8f O[4] = {};
    float mrow[8], lrow[8];
#pragma unroll
    for (int r = 0; r < 8; ++r) { mrow[r] = -3.0e38f; lrow[r] = 0.0f; }

    const int krow = tid >> 1;          // 0..63
    const int koff = (tid & 1) * 32;    // halves

    for (int jb = 0; jb < S; jb += 64) {
        // --- stage K block (row-major) and V block (transposed) into LDS ---
        {
            const uint4* gk = (const uint4*)(kp + (size_t)(jb + krow) * DK + koff);
            uint4* sk = (uint4*)(&Ks[krow][koff]);
            sk[0] = gk[0]; sk[1] = gk[1]; sk[2] = gk[2]; sk[3] = gk[3];
        }
        {
            const uint4* gv = (const uint4*)(vp + (size_t)(jb + krow) * DK + koff);
            uint4 tmp[4];
            tmp[0] = gv[0]; tmp[1] = gv[1]; tmp[2] = gv[2]; tmp[3] = gv[3];
            const h16* th = (const h16*)tmp;
#pragma unroll
            for (int dd = 0; dd < 32; ++dd)
                Vt[koff + dd][krow] = th[dd];
        }
        if (tid < 127) {  // bias slice: idx = i + 511 - j, u = (i-qb)-(j-jb)+63
            int idx = (qb - jb + 448) + tid;
            sBias[tid] = rel_table[idx * H + h];
        }
        __syncthreads();

        // --- scores: S = Q * K^T (dk=64 -> 2 K-chunks) ---
        v8f sc[4] = {};
#pragma unroll
        for (int t = 0; t < 4; ++t) {
#pragma unroll
            for (int c = 0; c < 2; ++c) {
                v16h kf = load_frag_B(&Ks[t * 16 + lo][c * 32], hi);
                sc[t] = WMMA_F16(qf[c], kf, sc[t]);
            }
        }
        // --- add relative-position bias ---
#pragma unroll
        for (int t = 0; t < 4; ++t)
#pragma unroll
            for (int r = 0; r < 8; ++r)
                sc[t][r] += sBias[w * 16 + r + 8 * hi - t * 16 - lo + 63];

        // --- online softmax (row stats via half-wave xor reductions) ---
        float alpha[8];
#pragma unroll
        for (int r = 0; r < 8; ++r) {
            float vmax = fmaxf(fmaxf(sc[0][r], sc[1][r]), fmaxf(sc[2][r], sc[3][r]));
#pragma unroll
            for (int msk = 1; msk < 16; msk <<= 1)
                vmax = fmaxf(vmax, __shfl_xor(vmax, msk, 32));
            float mnew = fmaxf(mrow[r], vmax);
            alpha[r] = __expf(mrow[r] - mnew);
            float s = 0.0f;
#pragma unroll
            for (int t = 0; t < 4; ++t) {
                float p = __expf(sc[t][r] - mnew);
                sc[t][r] = p;
                s += p;
            }
#pragma unroll
            for (int msk = 1; msk < 16; msk <<= 1)
                s += __shfl_xor(s, msk, 32);
            lrow[r] = alpha[r] * lrow[r] + s;
            mrow[r] = mnew;
        }
        // --- rescale running O ---
#pragma unroll
        for (int t = 0; t < 4; ++t)
#pragma unroll
            for (int r = 0; r < 8; ++r)
                O[t][r] *= alpha[r];

        // --- re-layout P (C layout -> A layout) through per-wave LDS ---
#pragma unroll
        for (int t = 0; t < 4; ++t)
#pragma unroll
            for (int r = 0; r < 8; ++r)
                Ps[w][r + 8 * hi][t * 16 + lo] = (h16)sc[t][r];
        asm volatile("s_wait_dscnt 0" ::: "memory");
        v16h pf[2];
#pragma unroll
        for (int c = 0; c < 2; ++c)
            pf[c] = load_frag_A(&Ps[w][lo][c * 32], hi);

        // --- O += P * V ---
#pragma unroll
        for (int t = 0; t < 4; ++t) {
#pragma unroll
            for (int c = 0; c < 2; ++c) {
                v16h vf = load_frag_B(&Vt[t * 16 + lo][c * 32], hi);
                O[t] = WMMA_F16(pf[c], vf, O[t]);
            }
        }
        __syncthreads();
    }

    // --- normalize and store (f16, row-major [b*S+s][h*64+d]) ---
    float inv[8];
#pragma unroll
    for (int r = 0; r < 8; ++r) inv[r] = 1.0f / lrow[r];
#pragma unroll
    for (int t = 0; t < 4; ++t)
#pragma unroll
        for (int r = 0; r < 8; ++r) {
            int i = qb + w * 16 + r + 8 * hi;
            int d = t * 16 + lo;
            outA[((size_t)(b * S + i)) * 1024 + h * 64 + d] = (h16)(O[t][r] * inv[r]);
        }
}

// ---------------------------------------------------------------------------
extern "C" void kernel_launch(void* const* d_in, const int* in_sizes, int n_in,
                              void* d_out, int out_size, void* d_ws, size_t ws_size,
                              hipStream_t stream) {
    (void)in_sizes; (void)n_in; (void)out_size; (void)ws_size;
    const float* x   = (const float*)d_in[0];
    const float* Wq  = (const float*)d_in[1];
    const float* bq  = (const float*)d_in[2];
    const float* Wk  = (const float*)d_in[3];
    const float* bk  = (const float*)d_in[4];
    const float* Wv  = (const float*)d_in[5];
    const float* bv  = (const float*)d_in[6];
    const float* Wo  = (const float*)d_in[7];
    const float* bo  = (const float*)d_in[8];
    const float* rel = (const float*)d_in[9];
    float* out = (float*)d_out;

    const int B = 16, S = 512, D = 1024, H = 16;
    const size_t nX = (size_t)B * S * D;   // 8M elems
    const size_t nW = (size_t)D * D;       // 1M elems

    char* ws = (char*)d_ws;
    h16* xh  = (h16*)ws; ws += nX * 2;
    h16* wqh = (h16*)ws; ws += nW * 2;
    h16* wkh = (h16*)ws; ws += nW * 2;
    h16* wvh = (h16*)ws; ws += nW * 2;
    h16* woh = (h16*)ws; ws += nW * 2;
    h16* qh  = (h16*)ws; ws += nX * 2;
    h16* kh  = (h16*)ws; ws += nX * 2;
    h16* vh  = (h16*)ws; ws += nX * 2;
    h16* ah  = (h16*)ws; ws += nX * 2;   // ~88 MB total

    const int thr = 256;
    f32_to_f16_kernel<<<(int)((nX / 8 + thr - 1) / thr), thr, 0, stream>>>(x,  xh,  (int)nX);
    f32_to_f16_kernel<<<(int)((nW / 8 + thr - 1) / thr), thr, 0, stream>>>(Wq, wqh, (int)nW);
    f32_to_f16_kernel<<<(int)((nW / 8 + thr - 1) / thr), thr, 0, stream>>>(Wk, wkh, (int)nW);
    f32_to_f16_kernel<<<(int)((nW / 8 + thr - 1) / thr), thr, 0, stream>>>(Wv, wvh, (int)nW);
    f32_to_f16_kernel<<<(int)((nW / 8 + thr - 1) / thr), thr, 0, stream>>>(Wo, woh, (int)nW);

    dim3 gg(D / 128, (B * S) / 128);   // (8, 64)
    // Q projection carries the 1/sqrt(dk) scale
    gemm_xwT_kernel<<<gg, 256, 0, stream>>>(xh, wqh, bq, qh, nullptr, B * S, D, D, 0, 0.125f);
    gemm_xwT_kernel<<<gg, 256, 0, stream>>>(xh, wkh, bk, kh, nullptr, B * S, D, D, 0, 1.0f);
    gemm_xwT_kernel<<<gg, 256, 0, stream>>>(xh, wvh, bv, vh, nullptr, B * S, D, D, 0, 1.0f);

    dim3 ga(S / 64, H, B);             // (8, 16, 16)
    attn_kernel<<<ga, 128, 0, stream>>>(qh, kh, vh, rel, ah);

    gemm_xwT_kernel<<<gg, 256, 0, stream>>>(ah, woh, bo, nullptr, out, B * S, D, D, 1, 1.0f);
}
